// SelfAttention_72335839199954
// MI455X (gfx1250) — matrix operand order repaired and checked by reading, not compiled, not run on hardware
//
#include <hip/hip_runtime.h>
#include <hip/hip_bf16.h>

// ---------------------------------------------------------------------------
// Self-attention for B=2, L=2048, D=1024, H=16, HS=64 on gfx1250 (MI455X).
// bf16 WMMA for all matmuls, flash-style softmax, double-buffered LDS tiles
// filled via the CDNA5 async-to-LDS path (ASYNCcnt).  M-blocked waves: every
// B-fragment read from LDS feeds two WMMAs.
// ---------------------------------------------------------------------------

#define B_  2
#define L_  2048
#define D_  1024
#define H_  16
#define HS_ 64

typedef __attribute__((ext_vector_type(16))) __bf16         v16bf;
typedef __attribute__((ext_vector_type(8)))  float          v8f;
typedef __attribute__((ext_vector_type(8)))  unsigned short v8us;
typedef __attribute__((ext_vector_type(16))) unsigned short v16us;
typedef __attribute__((ext_vector_type(4)))  int            v4i;
typedef unsigned short us;

#if __has_builtin(__builtin_amdgcn_global_load_async_to_lds_b128)
#define ASYNC_OK 1
#else
#define ASYNC_OK 0
#endif

// 16-byte global -> LDS copy (async DMA if available, sync fallback)
__device__ __forceinline__ void cp16(us* ldst, const us* gsrc) {
#if ASYNC_OK
    __builtin_amdgcn_global_load_async_to_lds_b128(
        (__attribute__((address_space(1))) v4i*)gsrc,
        (__attribute__((address_space(3))) v4i*)ldst,
        0, 0);
#else
    *(v8us*)ldst = *(const v8us*)gsrc;
#endif
}

__device__ __forceinline__ void wait_async() {
#if ASYNC_OK
#if __has_builtin(__builtin_amdgcn_s_wait_asynccnt)
    __builtin_amdgcn_s_wait_asynccnt(0);
#else
    asm volatile("s_wait_asynccnt 0x0" ::: "memory");
#endif
#endif
}

// float -> bf16, round-to-nearest-even
__device__ __forceinline__ us f2bf(float f) {
    unsigned int u = __float_as_uint(f);
    unsigned int r = u + 0x7FFFu + ((u >> 16) & 1u);
    return (us)(r >> 16);
}

__device__ __forceinline__ unsigned int pack2(float a, float b) {
    return (unsigned int)f2bf(a) | ((unsigned int)f2bf(b) << 16);
}

// A-fragment (16x32, 16-bit): lanes 0-15 hold K{0..7,16..23}; lanes 16-31 hold
// K{8..15,24..31}.  p = &row[m][k0] (16B aligned chunks).
__device__ __forceinline__ v16bf load_fragA(const us* p, int hi) {
    union { v16us v; v8us h[2]; } u;
    u.h[0] = *(const v8us*)(p + hi * 8);
    u.h[1] = *(const v8us*)(p + 16 + hi * 8);
    return __builtin_bit_cast(v16bf, u.v);
}

// B-fragment (32x16, 16-bit): lane n=l%16 holds K{8*hi..8*hi+7, 16+8*hi..16+8*hi+7}.
// p = &row[n][k_base]; works for both global and LDS pointers.
__device__ __forceinline__ v16bf load_fragB(const us* p, int hi) {
    union { v16us v; v8us h[2]; } u;
    u.h[0] = *(const v8us*)(p + hi * 8);
    u.h[1] = *(const v8us*)(p + 16 + hi * 8);
    return __builtin_bit_cast(v16bf, u.v);
}

__device__ __forceinline__ v8f wmma_bf16(v16bf a, v16bf b, v8f c) {
    // (neg_a, A, neg_b, B, c_mod, C, reuse_a, reuse_b)
    return __builtin_amdgcn_wmma_f32_16x16x32_bf16(false, a, false, b,
                                                   (short)0, c, false, false);
}

// wave32 xor-shuffle via LDS permute hardware (no LDS memory touched)
__device__ __forceinline__ float shfl_xor_f(float v, int m) {
    int idx = (((int)(threadIdx.x & 31u)) ^ m) << 2;
    return __int_as_float(__builtin_amdgcn_ds_bpermute(idx, __float_as_int(v)));
}

// ---------------------------------------------------------------------------
// fp32 -> bf16 conversion (x and the three weight matrices)
// ---------------------------------------------------------------------------
__global__ void cvt_bf16_kernel(const float* __restrict__ in,
                                us* __restrict__ out, int n) {
    int i = blockIdx.x * blockDim.x + threadIdx.x;
    int stride = gridDim.x * blockDim.x;
    for (; i < n; i += stride)
        out[i] = f2bf(__builtin_nontemporal_load(in + i));
}

// ---------------------------------------------------------------------------
// Projection GEMM:  Out = X(bf16)[4096,1024] @ W(bf16)[1024,1024]^T + bias
// Output written bf16 either as [B,H,L,HS] (Q,K) or [B,H,HS,L] (V^T).
// block = 256 threads (8 waves); wave tile = 32(M) x 64(N); block = 256 x 64.
// W tile (64n x 32k) double-buffered in LDS, shared by all 8 waves; every
// B-fragment feeds two WMMAs (M-blocking).
// ---------------------------------------------------------------------------
__global__ __launch_bounds__(256)
void qkv_gemm_kernel(const us* __restrict__ Xb, const us* __restrict__ Wb,
                     const float* __restrict__ bias, us* __restrict__ Out,
                     int transposed) {
    const int tid  = threadIdx.x;
    const int w    = tid >> 5;
    const int lane = tid & 31;
    const int li   = lane & 15;
    const int hi   = lane >> 4;

    const int nbase = blockIdx.x * 64;
    const int mwave = blockIdx.y * 256 + w * 32;

    __shared__ __align__(16) us ldsW[2][64 * 32];   // 2 x 4KB

    // cooperative staging: each of 256 threads owns one 16B chunk of the tile
    const int srow = tid >> 2;            // 0..63  (n within tile)
    const int scol = (tid & 3) * 8;       // 0,8,16,24 (k within tile)
    const us* wsrc = Wb + (size_t)(nbase + srow) * D_ + scol;
    us*       wdst0 = &ldsW[0][srow * 32 + scol];
    us*       wdst1 = &ldsW[1][srow * 32 + scol];

    const us* Arow0 = Xb + (size_t)(mwave + li) * D_;
    const us* Arow1 = Xb + (size_t)(mwave + 16 + li) * D_;

    // prologue: stage tile 0, preload A frags 0
    cp16(wdst0, wsrc);
    v16bf a_cur0 = load_fragA(Arow0, hi);
    v16bf a_cur1 = load_fragA(Arow1, hi);
    wait_async();
    __syncthreads();

    v8f acc[2][4] = {};

    for (int k0 = 0; k0 < D_; k0 += 32) {
        const int cur = (k0 >> 5) & 1;
        v16bf a_nxt0 = a_cur0, a_nxt1 = a_cur1;
        if (k0 + 32 < D_) {
            cp16(cur ? wdst0 : wdst1, wsrc + k0 + 32);   // overlap with math
            a_nxt0 = load_fragA(Arow0 + k0 + 32, hi);
            a_nxt1 = load_fragA(Arow1 + k0 + 32, hi);
        }
        const us* lw = &ldsW[cur][0];
#pragma unroll
        for (int t = 0; t < 4; ++t) {
            const v16bf bfr = load_fragB(lw + (t * 16 + li) * 32, hi);
            acc[0][t] = wmma_bf16(a_cur0, bfr, acc[0][t]);
            acc[1][t] = wmma_bf16(a_cur1, bfr, acc[1][t]);
        }
        a_cur0 = a_nxt0;
        a_cur1 = a_nxt1;
        wait_async();
        __syncthreads();
    }

#pragma unroll
    for (int mt = 0; mt < 2; ++mt) {
#pragma unroll
        for (int t = 0; t < 4; ++t) {
            const int n  = nbase + t * 16 + li;   // output feature (C-frag col)
            const float bv = bias[n];
            const int h  = n >> 6;                // head
            const int hs = n & 63;                // within-head dim
#pragma unroll
            for (int r = 0; r < 8; ++r) {
                const int m    = mwave + mt * 16 + r + 8 * hi;  // token
                const int b    = m >> 11;                        // m / L_
                const int ltok = m & (L_ - 1);
                const us  o    = f2bf(acc[mt][t][r] + bv);
                if (!transposed)
                    Out[((size_t)(b * H_ + h) * L_ + ltok) * HS_ + hs] = o;  // [B,H,L,HS]
                else
                    Out[((size_t)(b * H_ + h) * HS_ + hs) * L_ + ltok] = o;  // [B,H,HS,L]
            }
        }
    }
}

// ---------------------------------------------------------------------------
// Flash attention: one block = 4 waves = 128 query rows of one (b,h); each
// wave owns two 16-row M-tiles so every K/V B-fragment feeds two WMMAs.
// K tile (32 tok x 64 hs) and V^T tile (64 hs x 32 tok) double-buffered in
// LDS, shared by the 4 waves; async-prefetched one iteration ahead.
// ---------------------------------------------------------------------------
__global__ __launch_bounds__(128)
void attn_kernel(const us* __restrict__ Qb, const us* __restrict__ Kb,
                 const us* __restrict__ Vt, const float* __restrict__ mask,
                 float* __restrict__ out) {
    const int tid  = threadIdx.x;
    const int w    = tid >> 5;
    const int lane = tid & 31;
    const int li   = lane & 15;
    const int hi   = lane >> 4;

    const int bx   = blockIdx.x;
    const int qblk = bx & 15;           // L/128 = 16 query blocks
    const int h    = (bx >> 4) & 15;
    const int b    = bx >> 8;
    const int bh   = b * H_ + h;
    const int qbase = qblk * 128 + w * 32;

    __shared__ __align__(16) us ldsK[2][32 * 64];   // 2 x 4KB
    __shared__ __align__(16) us ldsV[2][64 * 32];   // 2 x 4KB
    __shared__ __align__(16) us Pl[4][1024];        // per-wave P staging (2 tiles)
    us* Pw = &Pl[w][0];

    // cooperative staging: 128 threads x 2 chunks per tile
    // K tile: 32 rows x 128B ; V tile: 64 rows x 64B  -> 256 x 16B chunks each
    const int c0 = tid * 2;
    const int kr0 = c0 >> 2,       kc0 = (c0 & 3) * 8;
    const int kr1 = (c0 + 1) >> 2, kc1 = ((c0 + 1) & 3) * 8;
    const us* ksrc0 = Kb + (size_t)(bh * L_ + kr0) * HS_ + kc0;
    const us* ksrc1 = Kb + (size_t)(bh * L_ + kr1) * HS_ + kc1;
    const us* vsrc0 = Vt + ((size_t)bh * HS_ + kr0) * L_ + kc0;
    const us* vsrc1 = Vt + ((size_t)bh * HS_ + kr1) * L_ + kc1;

    // Q fragments for this wave's 2 x 16 rows (resident for whole kernel)
    v16bf Aq[2][2];
#pragma unroll
    for (int mt = 0; mt < 2; ++mt) {
        const us* Qrow = Qb + (size_t)(bh * L_ + qbase + mt * 16 + li) * HS_;
        Aq[mt][0] = load_fragA(Qrow + 0,  hi);   // hs 0..31
        Aq[mt][1] = load_fragA(Qrow + 32, hi);   // hs 32..63
    }

    v8f o[2][4] = {};
    float mi[2][8], lsum[2][8];
#pragma unroll
    for (int mt = 0; mt < 2; ++mt)
#pragma unroll
        for (int r = 0; r < 8; ++r) { mi[mt][r] = -3.0e38f; lsum[mt][r] = 0.0f; }

    const float scale = 0.125f;  // 1/sqrt(64)

    // prologue: stage tiles for kb = 0
    cp16(&ldsK[0][kr0 * 64 + kc0], ksrc0);
    cp16(&ldsK[0][kr1 * 64 + kc1], ksrc1);
    cp16(&ldsV[0][kr0 * 32 + kc0], vsrc0);
    cp16(&ldsV[0][kr1 * 32 + kc1], vsrc1);
    wait_async();
    __syncthreads();

    for (int kb = 0; kb < L_; kb += 32) {
        const int cur = (kb >> 5) & 1;
        const int nxt = cur ^ 1;

        // prefetch next K/V tiles into the other buffer (overlaps with math)
        if (kb + 32 < L_) {
            cp16(&ldsK[nxt][kr0 * 64 + kc0], ksrc0 + (size_t)(kb + 32) * HS_);
            cp16(&ldsK[nxt][kr1 * 64 + kc1], ksrc1 + (size_t)(kb + 32) * HS_);
            cp16(&ldsV[nxt][kr0 * 32 + kc0], vsrc0 + kb + 32);
            cp16(&ldsV[nxt][kr1 * 32 + kc1], vsrc1 + kb + 32);
        }

        // ---- scores: S = Q K^T from LDS; B-frags shared by both M-tiles ----
        const us* lk = &ldsK[cur][0];
        const v16bf Bk00 = load_fragB(lk + (li)      * 64 + 0,  hi);
        const v16bf Bk01 = load_fragB(lk + (li)      * 64 + 32, hi);
        const v16bf Bk10 = load_fragB(lk + (li + 16) * 64 + 0,  hi);
        const v16bf Bk11 = load_fragB(lk + (li + 16) * 64 + 32, hi);
        v8f s0[2], s1[2];
#pragma unroll
        for (int mt = 0; mt < 2; ++mt) {
            v8f z0 = {}, z1 = {};
            z0 = wmma_bf16(Aq[mt][0], Bk00, z0);
            z0 = wmma_bf16(Aq[mt][1], Bk01, z0);
            z1 = wmma_bf16(Aq[mt][0], Bk10, z1);
            z1 = wmma_bf16(Aq[mt][1], Bk11, z1);
            s0[mt] = z0; s1[mt] = z1;
        }

        // ---- online softmax + P staging per M-tile ----
        v16bf Pf[2];
#pragma unroll
        for (int mt = 0; mt < 2; ++mt) {
            float corr[8], p0a[8], p1a[8];
#pragma unroll
            for (int r = 0; r < 8; ++r) {
                const int q = qbase + mt * 16 + r + 8 * hi;
                const float* mp = mask + ((size_t)b * L_ + q) * (size_t)L_ + kb;
                float v0 = s0[mt][r] * scale + __builtin_nontemporal_load(mp + li);
                float v1 = s1[mt][r] * scale + __builtin_nontemporal_load(mp + 16 + li);
                float mx = fmaxf(v0, v1);
#pragma unroll
                for (int off = 1; off < 16; off <<= 1)
                    mx = fmaxf(mx, shfl_xor_f(mx, off));
                const float mnew = fmaxf(mi[mt][r], mx);
                const float c    = __expf(mi[mt][r] - mnew);
                const float e0   = __expf(v0 - mnew);
                const float e1   = __expf(v1 - mnew);
                float rs = e0 + e1;
#pragma unroll
                for (int off = 1; off < 16; off <<= 1)
                    rs += shfl_xor_f(rs, off);
                lsum[mt][r] = lsum[mt][r] * c + rs;
                mi[mt][r]   = mnew;
                corr[r] = c;
                p0a[r]  = e0;
                p1a[r]  = e1;
            }
#pragma unroll
            for (int r = 0; r < 8; ++r) {
                o[mt][0][r] *= corr[r]; o[mt][1][r] *= corr[r];
                o[mt][2][r] *= corr[r]; o[mt][3][r] *= corr[r];
            }

            // C-layout -> A-layout for P via per-wave LDS (col-major)
            us* Pm = Pw + mt * 512;
            uint4 w0, w1;
            w0.x = pack2(p0a[0], p0a[1]); w0.y = pack2(p0a[2], p0a[3]);
            w0.z = pack2(p0a[4], p0a[5]); w0.w = pack2(p0a[6], p0a[7]);
            w1.x = pack2(p1a[0], p1a[1]); w1.y = pack2(p1a[2], p1a[3]);
            w1.z = pack2(p1a[4], p1a[5]); w1.w = pack2(p1a[6], p1a[7]);
            *(uint4*)&Pm[(li)      * 16 + 8 * hi] = w0;   // cols 0..15
            *(uint4*)&Pm[(li + 16) * 16 + 8 * hi] = w1;   // cols 16..31

            union { v16us v; us e[16]; } uf;
#pragma unroll
            for (int i = 0; i < 16; ++i) {
                const int c = (i < 8) ? (8 * hi + i) : (16 + 8 * hi + (i - 8));
                uf.e[i] = Pm[c * 16 + li];
            }
            Pf[mt] = __builtin_bit_cast(v16bf, uf.v);
        }

        // ---- O += P @ V from LDS; V B-frags shared by both M-tiles ----
        const us* lv = &ldsV[cur][0];
#pragma unroll
        for (int t = 0; t < 4; ++t) {
            const v16bf bv = load_fragB(lv + (t * 16 + li) * 32, hi);
            o[0][t] = wmma_bf16(Pf[0], bv, o[0][t]);
            o[1][t] = wmma_bf16(Pf[1], bv, o[1][t]);
        }

        wait_async();
        __syncthreads();
    }

    // ---- epilogue: normalize and stream out fp32 [B,L,D] (non-temporal) ----
#pragma unroll
    for (int mt = 0; mt < 2; ++mt) {
#pragma unroll
        for (int r = 0; r < 8; ++r) {
            const int q = qbase + mt * 16 + r + 8 * hi;
            const float inv = 1.0f / lsum[mt][r];
            float* op = out + ((size_t)b * L_ + q) * D_ + h * HS_;
            __builtin_nontemporal_store(o[mt][0][r] * inv, op + 0  + li);
            __builtin_nontemporal_store(o[mt][1][r] * inv, op + 16 + li);
            __builtin_nontemporal_store(o[mt][2][r] * inv, op + 32 + li);
            __builtin_nontemporal_store(o[mt][3][r] * inv, op + 48 + li);
        }
    }
}

// ---------------------------------------------------------------------------
// Host launcher
// ---------------------------------------------------------------------------
extern "C" void kernel_launch(void* const* d_in, const int* in_sizes, int n_in,
                              void* d_out, int out_size, void* d_ws, size_t ws_size,
                              hipStream_t stream) {
    (void)in_sizes; (void)n_in; (void)out_size; (void)ws_size;

    const float* x    = (const float*)d_in[0];
    const float* mask = (const float*)d_in[1];
    const float* Wq   = (const float*)d_in[2];
    const float* bq   = (const float*)d_in[3];
    const float* Wk   = (const float*)d_in[4];
    const float* bk   = (const float*)d_in[5];
    const float* Wv   = (const float*)d_in[6];
    const float* bv   = (const float*)d_in[7];
    float* out        = (float*)d_out;

    const size_t XB_N = (size_t)B_ * L_ * D_;        // 4,194,304
    const size_t W_N  = (size_t)D_ * D_;             // 1,048,576
    const size_t Q_N  = (size_t)B_ * H_ * L_ * HS_;  // 4,194,304

    us* xb  = (us*)d_ws;
    us* wqb = xb  + XB_N;
    us* wkb = wqb + W_N;
    us* wvb = wkb + W_N;
    us* Qb  = wvb + W_N;
    us* Kb  = Qb  + Q_N;
    us* Vt  = Kb  + Q_N;

    // 1) fp32 -> bf16
    cvt_bf16_kernel<<<4096, 256, 0, stream>>>(x,  xb,  (int)XB_N);
    cvt_bf16_kernel<<<2048, 256, 0, stream>>>(Wq, wqb, (int)W_N);
    cvt_bf16_kernel<<<2048, 256, 0, stream>>>(Wk, wkb, (int)W_N);
    cvt_bf16_kernel<<<2048, 256, 0, stream>>>(Wv, wvb, (int)W_N);

    // 2) projections (Q, K row-major per head; V transposed per head)
    dim3 ggrid(D_ / 64, (B_ * L_) / 256);   // (16, 16)
    qkv_gemm_kernel<<<ggrid, 256, 0, stream>>>(xb, wqb, bq, Qb, 0);
    qkv_gemm_kernel<<<ggrid, 256, 0, stream>>>(xb, wkb, bk, Kb, 0);
    qkv_gemm_kernel<<<ggrid, 256, 0, stream>>>(xb, wvb, bv, Vt, 1);

    // 3) flash attention
    attn_kernel<<<B_ * H_ * (L_ / 128), 128, 0, stream>>>(Qb, Kb, Vt, mask, out);
}
